// Spconv1x3LeakyReLUBatchNorm_45861660786969
// MI455X (gfx1250) — matrix operand order repaired and checked by reading, not compiled, not run on hardware
//
#include <hip/hip_runtime.h>
#include <stdint.h>

// CDNA5 (gfx1250) wave32 WMMA types
typedef __attribute__((ext_vector_type(16))) _Float16 v16h;
typedef __attribute__((ext_vector_type(8)))  float    v8f;

#define KTAPS 9
#define CIN 32
#define COUT 32
#define NEG_SLOPE 0.01f
#define BN_EPS 1e-5f

// -------------------------------------------------------------------------
// Phase 0: zero the 32-float "zero page" used for masked-row redirects.
// -------------------------------------------------------------------------
__global__ void zero_page_kernel(float* __restrict__ zbuf)
{
    zbuf[threadIdx.x] = 0.0f;   // launched with 32 threads
}

// -------------------------------------------------------------------------
// Phase 1: rulebook gather -> f16 WMMA GEMM over 9 taps -> LeakyReLU
//          -> write pre-BN activations to d_out, per-block BN partial sums.
// One wave = one 32-voxel x 32-cout tile:
//   two A fragments (rows 0-15, 16-31) x two B fragments (cout halves)
//   -> four 16x16 f32 accumulators, 4 WMMAs per tap (B reused across A tiles).
// Masked / OOB rows redirect the gather base pointer to a zeroed scratch
// line: mask byte is loaded UNCONDITIONALLY from a clamped address and the
// select is done in registers (branch-free v_cndmask, no exec-mask churn),
// and masked lanes stop pulling random 128B rows through L2.
// Block = 256 threads = 8 waves = 256 voxels.
// -------------------------------------------------------------------------
__global__ __launch_bounds__(256) void spconv_leaky_wmma_kernel(
    const float* __restrict__ features,        // [N][32] f32
    const float* __restrict__ weight,          // [9][32][32] f32 (k, cin, cout)
    const int*   __restrict__ nidx,            // [9][N] i32
    const unsigned char* __restrict__ nmask,   // [9][N] bool (1 byte)
    const float* __restrict__ zbuf,            // [32] zeros
    float* __restrict__ y_out,                 // [N][32] pre-BN activations
    float* __restrict__ psum,                  // [nBlocks][32]
    float* __restrict__ psumsq,                // [nBlocks][32]
    int N)
{
    // B fragments in LDS, pre-converted to f16 in wave-fragment order:
    // entry index = ((k*2 + h)*32 + lane)*16 + j
    //   lane 0-15 : K = 0..7, 16..23 (j<8: j, j>=8: 8+j),  cout = h*16 + (lane&15)
    //   lane16-31 : K = 8..15, 24..31 (j<8: 8+j, j>=8: 16+j), cout = h*16 + (lane&15)
    __shared__ alignas(32) _Float16 wlds[KTAPS * 2 * 32 * 16];   // 18 KB
    __shared__ float wsum  [8 * 64];   // [wave][group*32 + channel]
    __shared__ float wsumsq[8 * 64];

    const int tid = threadIdx.x;

    // Convert 9x32x32 f32 weights into the LDS B-fragment layout.
    for (int e = tid; e < KTAPS * 2 * 32 * 16; e += 256) {
        int k   = e >> 10;                 // / 1024
        int rem = e & 1023;
        int h   = rem >> 9;                // cout half
        int l   = (rem >> 4) & 31;         // lane
        int j   = e & 15;
        int kin  = ((l >> 4) << 3) + (j & 7) + ((j >> 3) << 4);    // table order: 8g+j / 16+8g+(j-8)
        int cout = (h << 4) + (l & 15);
        wlds[e] = (_Float16)weight[(k * CIN + kin) * COUT + cout];
    }
    __syncthreads();

    const int wave = tid >> 5;
    const int lane = tid & 31;
    const int g    = lane >> 4;            // lane group (0: K-low, 1: K-high)
    const int ln   = lane & 15;            // row-within-tile / cout lane
    const int tileBase = (blockIdx.x * 8 + wave) * 32;

    v8f acc00 = {};  // rows 0-15,  cout 0..15
    v8f acc01 = {};  // rows 0-15,  cout 16..31
    v8f acc10 = {};  // rows 16-31, cout 0..15
    v8f acc11 = {};  // rows 16-31, cout 16..31
    float s0 = 0.f, s1 = 0.f, ss0 = 0.f, ss1 = 0.f;

    if (tileBase < N) {
        // A-fragment gather columns for this lane (16-bit A 16x32 layout):
        //   lane<16 : K = 0..7  and 16..23 of row M=lane
        //   lane>=16: K = 8..15 and 24..31 of row M=lane-16
        const int colBase = g * 8;
        const v16h* wl = (const v16h*)wlds;

        const int r0  = tileBase + ln;           // row in tile 0
        const int r1  = r0 + 16;                 // row in tile 1
        const bool in0 = (r0 < N);
        const bool in1 = (r1 < N);
        const int c0  = in0 ? r0 : (N - 1);      // clamped (always-valid addr)
        const int c1  = in1 ? r1 : (N - 1);

        #pragma unroll
        for (int k = 0; k < KTAPS; ++k) {
            const int base = k * N;
            const int i0 = base + c0;
            const int i1 = base + c1;
            // Unconditional clamped loads -> no exec-mask predication.
            int row0 = nidx[i0];
            int row1 = nidx[i1];
            int mb0  = (int)nmask[i0];
            int mb1  = (int)nmask[i1];
            // Branch-free combine (non-short-circuit '&').
            bool use0 = (mb0 != 0) & in0;
            bool use1 = (mb1 != 0) & in1;

            // Zero-page redirect: masked/OOB rows read the shared zero line.
            const float* fp0 =
                (use0 ? (features + (size_t)row0 * CIN) : zbuf) + colBase;
            const float* fp1 =
                (use1 ? (features + (size_t)row1 * CIN) : zbuf) + colBase;

            float4 p0 = ((const float4*)fp0)[0];
            float4 p1 = ((const float4*)fp0)[1];
            float4 p2 = ((const float4*)(fp0 + 16))[0];
            float4 p3 = ((const float4*)(fp0 + 16))[1];
            float4 q0 = ((const float4*)fp1)[0];
            float4 q1 = ((const float4*)fp1)[1];
            float4 q2 = ((const float4*)(fp1 + 16))[0];
            float4 q3 = ((const float4*)(fp1 + 16))[1];

            v16h a0, a1;
            a0[0]  = (_Float16)p0.x; a0[1]  = (_Float16)p0.y;
            a0[2]  = (_Float16)p0.z; a0[3]  = (_Float16)p0.w;
            a0[4]  = (_Float16)p1.x; a0[5]  = (_Float16)p1.y;
            a0[6]  = (_Float16)p1.z; a0[7]  = (_Float16)p1.w;
            a0[8]  = (_Float16)p2.x; a0[9]  = (_Float16)p2.y;
            a0[10] = (_Float16)p2.z; a0[11] = (_Float16)p2.w;
            a0[12] = (_Float16)p3.x; a0[13] = (_Float16)p3.y;
            a0[14] = (_Float16)p3.z; a0[15] = (_Float16)p3.w;

            a1[0]  = (_Float16)q0.x; a1[1]  = (_Float16)q0.y;
            a1[2]  = (_Float16)q0.z; a1[3]  = (_Float16)q0.w;
            a1[4]  = (_Float16)q1.x; a1[5]  = (_Float16)q1.y;
            a1[6]  = (_Float16)q1.z; a1[7]  = (_Float16)q1.w;
            a1[8]  = (_Float16)q2.x; a1[9]  = (_Float16)q2.y;
            a1[10] = (_Float16)q2.z; a1[11] = (_Float16)q2.w;
            a1[12] = (_Float16)q3.x; a1[13] = (_Float16)q3.y;
            a1[14] = (_Float16)q3.z; a1[15] = (_Float16)q3.w;

            v16h b0 = wl[(k * 2 + 0) * 32 + lane];
            v16h b1 = wl[(k * 2 + 1) * 32 + lane];
            acc00 = __builtin_amdgcn_wmma_f32_16x16x32_f16(
                false, a0, false, b0, (short)0, acc00, false, false);
            acc01 = __builtin_amdgcn_wmma_f32_16x16x32_f16(
                false, a0, false, b1, (short)0, acc01, false, false);
            acc10 = __builtin_amdgcn_wmma_f32_16x16x32_f16(
                false, a1, false, b0, (short)0, acc10, false, false);
            acc11 = __builtin_amdgcn_wmma_f32_16x16x32_f16(
                false, a1, false, b1, (short)0, acc11, false, false);
        }

        // C/D layout: VGPR r -> M = r + g*8, N = ln. LeakyReLU + store + partials.
        #pragma unroll
        for (int r = 0; r < 8; ++r) {
            int row = tileBase + r + g * 8;          // tile 0
            if (row < N) {
                float v  = acc00[r];
                float yv = v >= 0.0f ? v : NEG_SLOPE * v;
                y_out[(size_t)row * COUT + ln] = yv;
                s0 += yv; ss0 += yv * yv;

                v  = acc01[r];
                yv = v >= 0.0f ? v : NEG_SLOPE * v;
                y_out[(size_t)row * COUT + 16 + ln] = yv;
                s1 += yv; ss1 += yv * yv;
            }
            row += 16;                                // tile 1
            if (row < N) {
                float v  = acc10[r];
                float yv = v >= 0.0f ? v : NEG_SLOPE * v;
                y_out[(size_t)row * COUT + ln] = yv;
                s0 += yv; ss0 += yv * yv;

                v  = acc11[r];
                yv = v >= 0.0f ? v : NEG_SLOPE * v;
                y_out[(size_t)row * COUT + 16 + ln] = yv;
                s1 += yv; ss1 += yv * yv;
            }
        }
    }

    // Deterministic block reduction: each (wave, group, channel) slot is unique.
    wsum  [wave * 64 + g * 32 + ln]      = s0;
    wsum  [wave * 64 + g * 32 + 16 + ln] = s1;
    wsumsq[wave * 64 + g * 32 + ln]      = ss0;
    wsumsq[wave * 64 + g * 32 + 16 + ln] = ss1;
    __syncthreads();

    if (tid < 32) {
        float S = 0.f, SS = 0.f;
        #pragma unroll
        for (int w = 0; w < 8; ++w) {
            S  += wsum  [w * 64 + tid] + wsum  [w * 64 + 32 + tid];
            SS += wsumsq[w * 64 + tid] + wsumsq[w * 64 + 32 + tid];
        }
        psum  [(size_t)blockIdx.x * 32 + tid] = S;
        psumsq[(size_t)blockIdx.x * 32 + tid] = SS;
    }
}

// -------------------------------------------------------------------------
// Phase 2: fold block partials into per-channel BN scale/bias (1 wave).
// -------------------------------------------------------------------------
__global__ void bn_stats_kernel(const float* __restrict__ psum,
                                const float* __restrict__ psumsq,
                                const float* __restrict__ gamma,
                                const float* __restrict__ beta,
                                float* __restrict__ sb,   // [64]: scale | bias
                                int nBlocks, int N)
{
    int c = threadIdx.x;
    if (c < 32) {
        float S = 0.f, SS = 0.f;
        for (int b = 0; b < nBlocks; ++b) {
            S  += psum  [(size_t)b * 32 + c];
            SS += psumsq[(size_t)b * 32 + c];
        }
        float inv_n = 1.0f / (float)N;
        float mean  = S * inv_n;
        float var   = SS * inv_n - mean * mean;   // biased, matches torch BN
        var = var < 0.f ? 0.f : var;
        float scale = rsqrtf(var + BN_EPS) * gamma[c];
        sb[c]      = scale;
        sb[32 + c] = beta[c] - mean * scale;
    }
}

// -------------------------------------------------------------------------
// Phase 3: apply BN in place on d_out with float4 traffic.
// Element columns repeat every 8 float4 groups (32 channels / 4).
// -------------------------------------------------------------------------
__global__ __launch_bounds__(256) void bn_apply_kernel(
    float* __restrict__ y, const float* __restrict__ sb, int total4)
{
    int i = blockIdx.x * 256 + threadIdx.x;
    if (i < total4) {
        float4 v  = ((float4*)y)[i];
        float4 sc = ((const float4*)sb)[i & 7];
        float4 bi = ((const float4*)(sb + 32))[i & 7];
        v.x = v.x * sc.x + bi.x;
        v.y = v.y * sc.y + bi.y;
        v.z = v.z * sc.z + bi.z;
        v.w = v.w * sc.w + bi.w;
        ((float4*)y)[i] = v;
    }
}

extern "C" void kernel_launch(void* const* d_in, const int* in_sizes, int n_in,
                              void* d_out, int out_size, void* d_ws, size_t ws_size,
                              hipStream_t stream)
{
    const float* features = (const float*)d_in[0];           // [N,32]
    const float* weight   = (const float*)d_in[1];           // [9,32,32]
    const float* gamma    = (const float*)d_in[2];           // [32]
    const float* beta     = (const float*)d_in[3];           // [32]
    const int*   nidx     = (const int*)d_in[4];             // [9,N]
    const unsigned char* nmask = (const unsigned char*)d_in[5]; // [9,N] bool

    const int N = in_sizes[0] / CIN;
    const int nBlocks = (N + 255) / 256;                     // 256 voxels / block

    // Workspace layout (floats): psum | psumsq | sb[64] | zbuf[32]
    float* psum   = (float*)d_ws;
    float* psumsq = psum   + (size_t)nBlocks * 32;
    float* sb     = psumsq + (size_t)nBlocks * 32;
    float* zbuf   = sb + 64;           // 256B-aligned (offsets are x128B)
    float* y      = (float*)d_out;

    zero_page_kernel<<<1, 32, 0, stream>>>(zbuf);

    spconv_leaky_wmma_kernel<<<nBlocks, 256, 0, stream>>>(
        features, weight, nidx, nmask, zbuf, y, psum, psumsq, N);

    bn_stats_kernel<<<1, 32, 0, stream>>>(psum, psumsq, gamma, beta, sb, nBlocks, N);

    const int total4 = (N * COUT) / 4;
    bn_apply_kernel<<<(total4 + 255) / 256, 256, 0, stream>>>(y, sb, total4);
}
